// LuongAttention_74174085202239
// MI455X (gfx1250) — matrix operand order repaired and checked by reading, not compiled, not run on hardware
//
#include <hip/hip_runtime.h>

// MI455X / gfx1250, wave32. Fused Luong attention:
//   scores = enc @ dec^T  (softmax over dec-time)
//   context = softmax(scores) @ enc
//   out = tanh([context, dec] @ W_tanh)
// Big contractions run on v_wmma_f32_16x16x32_bf16 (f32 accumulate).

typedef __attribute__((ext_vector_type(16))) __bf16 v16bf;
typedef __attribute__((ext_vector_type(8)))  __bf16 v8bf;
typedef __attribute__((ext_vector_type(8)))  float  v8f;

static constexpr int kB = 8;
static constexpr int kS = 2048;
static constexpr int kD = 256;

// ---------------------------------------------------------------------------
// Kernel 1: flash-style attention. One wave per 16 enc(query) rows.
// Block = 256 threads = 8 waves; all 8 waves share one batch, so the 32-row
// dec/enc tiles per iteration are staged cooperatively into LDS (bf16).
// Writes normalized context [B,S,D] as bf16 into d_ws.
// ---------------------------------------------------------------------------
__global__ __launch_bounds__(256) void attn_fused_kernel(
    const float* __restrict__ enc, const float* __restrict__ dec,
    __bf16* __restrict__ ctxOut)
{
    __shared__ __align__(16) __bf16 sDec[32 * kD];   // [t][d]   (scores B-frags)
    __shared__ __align__(16) __bf16 sEncT[kD * 32];  // [d][t]   (context B-frags)
    __shared__ __align__(16) __bf16 sP[8][16 * 32];  // per-wave P tile [s][t]

    const int tid  = threadIdx.x;
    const int lane = tid & 31;
    const int wid  = tid >> 5;
    const int hf   = lane >> 4;   // lane half (0: lanes 0-15, 1: lanes 16-31)
    const int ln   = lane & 15;

    const int gwave = blockIdx.x * 8 + wid;   // 0..1023
    const int b  = gwave >> 7;                // kS/16 = 128 q-tiles per batch
    const int s0 = (gwave & 127) * 16;        // first query row of this wave

    const float* encB = enc + (size_t)b * kS * kD;
    const float* decB = dec + (size_t)b * kS * kD;

    // --- Q A-fragments: 8 chunks of K=32 over D, held in registers ---------
    // A layout (16-bit, 16x32): lane = M, element e -> K = e + 8*hf (e<8)
    //                                              K = e + 8 + 8*hf (e>=8)
    v16bf qa[8];
    {
        const float* qrow = encB + (size_t)(s0 + ln) * kD;
        for (int c = 0; c < 8; ++c)
            for (int e = 0; e < 16; ++e) {
                int k = (e < 8) ? (e + 8 * hf) : (e + 8 + 8 * hf);
                qa[c][e] = (__bf16)qrow[c * 32 + k];
            }
    }

    // Context accumulators: 16 x (16x16 f32 C-tiles) = 16 rows x 256 cols.
    v8f acc[16];
    for (int i = 0; i < 16; ++i)
        for (int e = 0; e < 8; ++e) acc[i][e] = 0.0f;
    float mrow[8], lrow[8];   // online-softmax state for this half's 8 rows
    for (int r = 0; r < 8; ++r) { mrow[r] = -1e30f; lrow[r] = 0.0f; }

    __bf16* myP = sP[wid];

    for (int it = 0; it < kS / 32; ++it) {
        const int t0 = it * 32;
        __syncthreads();
        // Cooperative bf16 staging of 32-row dec tile and transposed enc tile.
        for (int i = tid; i < 32 * kD; i += 256) {
            int row = i >> 8;        // kD == 256
            int col = i & 255;
            sDec[row * kD + col]  = (__bf16)decB[(size_t)(t0 + row) * kD + col];
            sEncT[col * 32 + row] = (__bf16)encB[(size_t)(t0 + row) * kD + col];
        }
        if (it + 1 < kS / 32) {      // global_prefetch_b8 next tile
            __builtin_prefetch(decB + (size_t)(t0 + 32) * kD + tid * 32, 0, 1);
            __builtin_prefetch(encB + (size_t)(t0 + 32) * kD + tid * 32, 0, 1);
        }
        __syncthreads();

        // --- scores: two 16x16 tiles (t sub-tiles), K=256 in 8 WMMAs each --
        v8f s1, s2;
        for (int e = 0; e < 8; ++e) { s1[e] = 0.0f; s2[e] = 0.0f; }
        for (int c = 0; c < 8; ++c) {
            union { v16bf v; v8bf h[2]; } u;
            // B layout (32x16): lane = N(t), element e -> K(d) = 8*hf + e (e<8), 16 + 8*hf + (e-8) (e>=8)
            const __bf16* p0 = sDec + ln * kD + c * 32 + hf * 8;
            u.h[0] = *(const v8bf*)(p0);
            u.h[1] = *(const v8bf*)(p0 + 16);
            s1 = __builtin_amdgcn_wmma_f32_16x16x32_bf16(
                     false, qa[c], false, u.v, (short)0, s1, false, false);
            const __bf16* p1 = sDec + (16 + ln) * kD + c * 32 + hf * 8;
            u.h[0] = *(const v8bf*)(p1);
            u.h[1] = *(const v8bf*)(p1 + 16);
            s2 = __builtin_amdgcn_wmma_f32_16x16x32_bf16(
                     false, qa[c], false, u.v, (short)0, s2, false, false);
        }

        // --- online softmax over the 32 new columns ------------------------
        float sc[8];
        for (int r = 0; r < 8; ++r) {
            float tmax = fmaxf(s1[r], s2[r]);
            for (int d = 1; d < 16; d <<= 1)
                tmax = fmaxf(tmax, __shfl_xor(tmax, d, 32));
            float mnew  = fmaxf(mrow[r], tmax);
            float scale = __expf(mrow[r] - mnew);
            float p1 = __expf(s1[r] - mnew);
            float p2 = __expf(s2[r] - mnew);
            s1[r] = p1; s2[r] = p2;
            float ps = p1 + p2;
            for (int d = 1; d < 16; d <<= 1)
                ps += __shfl_xor(ps, d, 32);
            lrow[r] = lrow[r] * scale + ps;
            mrow[r] = mnew;
            sc[r] = scale;
        }
        for (int i = 0; i < 16; ++i)
            for (int r = 0; r < 8; ++r) acc[i][r] *= sc[r];

        // --- P tile: C-layout -> A-layout via per-wave LDS transpose -------
        for (int r = 0; r < 8; ++r) {
            myP[(r + 8 * hf) * 32 + ln]      = (__bf16)s1[r];
            myP[(r + 8 * hf) * 32 + 16 + ln] = (__bf16)s2[r];
        }
        union { v16bf v; v8bf h[2]; } pa;
        const __bf16* pr = myP + ln * 32;
        pa.h[0] = *(const v8bf*)(pr + 8 * hf);
        pa.h[1] = *(const v8bf*)(pr + 16 + 8 * hf);

        // --- context += P(16x32) @ enc_tile(32x256), 16 WMMAs --------------
        for (int d16 = 0; d16 < 16; ++d16) {
            union { v16bf v; v8bf h[2]; } bv;
            const __bf16* ep = sEncT + (d16 * 16 + ln) * 32 + hf * 8;
            bv.h[0] = *(const v8bf*)(ep);
            bv.h[1] = *(const v8bf*)(ep + 16);
            acc[d16] = __builtin_amdgcn_wmma_f32_16x16x32_bf16(
                           false, pa.v, false, bv.v, (short)0, acc[d16],
                           false, false);
        }
    }

    // --- normalize and store context (bf16) --------------------------------
    for (int r = 0; r < 8; ++r) {
        float inv = 1.0f / lrow[r];
        int srow = s0 + r + 8 * hf;
        __bf16* orow = ctxOut + ((size_t)b * kS + srow) * kD;
        for (int d16 = 0; d16 < 16; ++d16)
            orow[d16 * 16 + ln] = (__bf16)(acc[d16][r] * inv);
    }
}

// ---------------------------------------------------------------------------
// Kernel 2: out = tanh([context | dec] @ W_tanh), i.e. [16384,512]@[512,256].
// One wave per 16x16 output tile; K=512 in 16 bf16 WMMA steps.
// ---------------------------------------------------------------------------
__global__ __launch_bounds__(256) void dense_tanh_kernel(
    const __bf16* __restrict__ ctx, const float* __restrict__ dec,
    const float* __restrict__ W, float* __restrict__ out)
{
    const int tid  = threadIdx.x;
    const int lane = tid & 31;
    const int wid  = tid >> 5;
    const int hf   = lane >> 4;
    const int ln   = lane & 15;

    const int gwave   = blockIdx.x * 8 + wid;  // 0..16383
    const int rowTile = gwave >> 4;            // 1024 row tiles of 16
    const int nt      = gwave & 15;            // 16 column tiles of 16
    const int g0      = rowTile * 16;

    v8f c;
    for (int e = 0; e < 8; ++e) c[e] = 0.0f;

    for (int kc = 0; kc < 16; ++kc) {
        v16bf a;
        const int row = g0 + ln;
        if (kc < 8) {  // context half (already bf16; contiguous 8-elem runs)
            union { v16bf v; v8bf h[2]; } u;
            const __bf16* ap = ctx + (size_t)row * kD + kc * 32;
            u.h[0] = *(const v8bf*)(ap + 8 * hf);
            u.h[1] = *(const v8bf*)(ap + 16 + 8 * hf);
            a = u.v;
        } else {       // dec half (convert f32 -> bf16)
            const float* ap = dec + (size_t)row * kD + (kc - 8) * 32;
            for (int e = 0; e < 16; ++e) {
                int k = (e < 8) ? (e + 8 * hf) : (e + 8 + 8 * hf);
                a[e] = (__bf16)ap[k];
            }
        }
        v16bf bw;      // W B-fragment: k = kc*32 + (e<8 ? 8*hf + e : 8 + 8*hf + e), col = nt*16 + ln
        for (int e = 0; e < 16; ++e) {
            int k = kc * 32 + ((e < 8) ? (e + 8 * hf) : (e + 8 + 8 * hf));
            bw[e] = (__bf16)W[(size_t)k * kD + nt * 16 + ln];
        }
        c = __builtin_amdgcn_wmma_f32_16x16x32_bf16(
                false, a, false, bw, (short)0, c, false, false);
    }

    for (int r = 0; r < 8; ++r) {
        int row = g0 + r + 8 * hf;
        out[(size_t)row * kD + nt * 16 + ln] = tanhf(c[r]);
    }
}

// ---------------------------------------------------------------------------
extern "C" void kernel_launch(void* const* d_in, const int* in_sizes, int n_in,
                              void* d_out, int out_size, void* d_ws, size_t ws_size,
                              hipStream_t stream)
{
    const float* enc = (const float*)d_in[0];  // [B,S,D] f32
    const float* dec = (const float*)d_in[1];  // [B,S,D] f32
    const float* W   = (const float*)d_in[2];  // [2D,D] f32

    // Scratch: context [B,S,D] in bf16 = 8 MiB.
    __bf16* ctx = (__bf16*)d_ws;

    attn_fused_kernel<<<kB * (kS / 16) / 8, 256, 0, stream>>>(enc, dec, ctx);
    dense_tanh_kernel<<<(kB * kS / 16) * (kD / 16) / 8, 256, 0, stream>>>(
        ctx, dec, W, (float*)d_out);
}
